// PointSAMAttention_40973988004763
// MI455X (gfx1250) — matrix operand order repaired and checked by reading, not compiled, not run on hardware
//
#include <hip/hip_runtime.h>
#include <cmath>

typedef __attribute__((ext_vector_type(16))) _Float16 v16h;
typedef __attribute__((ext_vector_type(8)))  _Float16 h8;
typedef __attribute__((ext_vector_type(4)))  _Float16 h4;
typedef __attribute__((ext_vector_type(2)))  __fp16   fp16x2;   // return type of cvt_pkrtz
typedef __attribute__((ext_vector_type(8)))  float    v8f;

#define B_   2
#define H_   16
#define L_   2048
#define D_   64
#define QT   128            // Q rows per block (8 waves x 16)
#define KT   64             // K rows per iteration
#define KSTR 72             // LDS row strides in halfs (144B = 36 dwords, 16B aligned)
#define VSTR 72
#define PSTR 72
#define NEG_BIG  (-1.0e9f)

union V16 { v16h v; h8 p[2]; _Float16 h[16]; };
union H4U { h4 v; fp16x2 p[2]; };

__device__ __forceinline__ float softplus_fast(float t) {
  // softplus(t) = max(t,0) + log(1 + exp(-|t|));  __expf/__logf -> v_exp_f32/v_log_f32
  float u = __expf(-__builtin_fabsf(t));
  return fmaxf(t, 0.0f) + __logf(1.0f + u);
}

__global__ __launch_bounds__(256) void PointSAMAttention_40973988004763_kernel(
    const float* __restrict__ Q, const float* __restrict__ K,
    const float* __restrict__ V, const int* __restrict__ Msk,
    float* __restrict__ Out, float* __restrict__ Scores)
{
  __shared__ __attribute__((aligned(16))) _Float16 sK [KT * KSTR];      // 9216 B
  __shared__ __attribute__((aligned(16))) _Float16 sVt[D_ * VSTR];      // 9216 B (V^T, chunk-swizzled)
  __shared__ __attribute__((aligned(16))) _Float16 sP [8 * 16 * PSTR];  // 18432 B (per-wave P strips)

  const int tid  = threadIdx.x;
  const int wave = tid >> 5;
  const int lane = tid & 31;
  const int lh   = lane & 15;
  const int hi   = lane >> 4;

  const int bh = blockIdx.y;           // b*H + h
  const int b  = bh >> 4;              // H_ == 16
  const int q0 = blockIdx.x * QT;
  const size_t rowbase = (size_t)bh * L_;

  // ---- persistent Q A-fragments (16x32 f16 x2), ISA A-layout ----
  const int qrow = q0 + wave * 16 + lh;
  const float* qp = Q + (rowbase + qrow) * D_;
  V16 qa0, qa1;
#pragma unroll
  for (int s = 0; s < 8; ++s) {
    qa0.h[s]     = (_Float16)qp[      hi * 8 + s];
    qa0.h[8 + s] = (_Float16)qp[16  + hi * 8 + s];
    qa1.h[s]     = (_Float16)qp[32  + hi * 8 + s];
    qa1.h[8 + s] = (_Float16)qp[48  + hi * 8 + s];
  }

  v8f oacc[4];
#pragma unroll
  for (int j = 0; j < 4; ++j)
#pragma unroll
    for (int r = 0; r < 8; ++r) oacc[j][r] = 0.f;

  const int crow = tid >> 2;           // 0..63 : staged key row
  const int c0   = (tid & 3) * 16;     // 0/16/32/48 : d-column chunk

  for (int kt = 0; kt < L_; kt += KT) {
    __syncthreads();                   // previous iteration done with sK/sVt
    {
      const float4* kg4 = (const float4*)(K + (rowbase + kt + crow) * D_ + c0);
      const float4* vg4 = (const float4*)(V + (rowbase + kt + crow) * D_ + c0);
      if (kt + KT < L_) {
        __builtin_prefetch((const float*)kg4 + (size_t)KT * D_, 0, 1);   // global_prefetch_b8
        __builtin_prefetch((const float*)vg4 + (size_t)KT * D_, 0, 1);
      }
      const int sw  = ((crow >> 4) ^ (c0 >> 4)) & 3;   // XOR chunk swizzle
      const int kin = crow & 15;
#pragma unroll
      for (int u = 0; u < 4; ++u) {
        float4 kf = kg4[u];
        H4U kh;
        kh.p[0] = __builtin_amdgcn_cvt_pkrtz(kf.x, kf.y);   // v_cvt_pk_rtz_f16_f32
        kh.p[1] = __builtin_amdgcn_cvt_pkrtz(kf.z, kf.w);
        *(h4*)&sK[crow * KSTR + c0 + 4 * u] = kh.v;         // K row-major f16
        float4 vf = vg4[u];
        H4U vh;
        vh.p[0] = __builtin_amdgcn_cvt_pkrtz(vf.x, vf.y);
        vh.p[1] = __builtin_amdgcn_cvt_pkrtz(vf.z, vf.w);
        const int d = c0 + 4 * u;                           // Vt[d][k], swizzled 16-half chunks
        sVt[(d + 0) * VSTR + sw * 16 + kin] = vh.v[0];
        sVt[(d + 1) * VSTR + sw * 16 + kin] = vh.v[1];
        sVt[(d + 2) * VSTR + sw * 16 + kin] = vh.v[2];
        sVt[(d + 3) * VSTR + sw * 16 + kin] = vh.v[3];
      }
    }
    __syncthreads();

    // ---- GEMM1: S = Q.K^T ; softplus ; emit scores ; stage P (f16) ----
#pragma unroll
    for (int j = 0; j < 4; ++j) {
      const _Float16* kr = &sK[(j * 16 + lh) * KSTR];  // B-fragment: lane = N = key row
      V16 kb0, kb1;
      kb0.p[0] = *(const h8*)(kr +      hi * 8);
      kb0.p[1] = *(const h8*)(kr + 16 + hi * 8);
      kb1.p[0] = *(const h8*)(kr + 32 + hi * 8);
      kb1.p[1] = *(const h8*)(kr + 48 + hi * 8);

      v8f s;
#pragma unroll
      for (int r = 0; r < 8; ++r) s[r] = 0.f;
      s = __builtin_amdgcn_wmma_f32_16x16x32_f16(false, qa0.v, false, kb0.v, (short)0, s, false, false);
      s = __builtin_amdgcn_wmma_f32_16x16x32_f16(false, qa1.v, false, kb1.v, (short)0, s, false, false);

      const int kg = kt + j * 16 + lh;
      const float madd = (Msk[(size_t)b * L_ + kg] != 0) ? NEG_BIG : 0.0f;
      // fold hi*8 rows into base -> per-r stores get constant immediate offsets (r*8192 B)
      float* scb = Scores + (rowbase + q0 + wave * 16 + hi * 8) * (size_t)L_ + kg;
      _Float16* prow = &sP[(wave * 16 + hi * 8) * PSTR + j * 16 + lh];
#pragma unroll
      for (int r = 0; r < 8; ++r) {
        float t  = fmaf(s[r], 0.125f, madd);                 // 1/sqrt(64)
        float sp = softplus_fast(t) * (1.0f / (float)L_);
        __builtin_nontemporal_store(sp, scb + (size_t)r * L_);  // streaming 512MB, keep L2 for K/V
        prow[r * PSTR] = (_Float16)sp;
      }
    }

    asm volatile("" ::: "memory");     // keep P stores before P fragment loads (LDS in-order per wave)

    // ---- GEMM2: O += P.V ----
    V16 pa0, pa1;
    {
      const _Float16* pr = &sP[(wave * 16 + lh) * PSTR];
      pa0.p[0] = *(const h8*)(pr + hi * 16);
      pa0.p[1] = *(const h8*)(pr + hi * 16 + 8);
      pa1.p[0] = *(const h8*)(pr + 32 + hi * 16);
      pa1.p[1] = *(const h8*)(pr + 32 + hi * 16 + 8);
    }
#pragma unroll
    for (int j = 0; j < 4; ++j) {
      const _Float16* vr = &sVt[(j * 16 + lh) * VSTR];  // lane = N = d column
      const int sw0 = (hi ^ j) & 3;                     // chunk (k0=0)  ^ (d>>4)
      const int sw1 = ((2 + hi) ^ j) & 3;               // chunk (k0=32) ^ (d>>4)
      V16 vb0, vb1;
      vb0.p[0] = *(const h8*)(vr + sw0 * 16);
      vb0.p[1] = *(const h8*)(vr + sw0 * 16 + 8);
      vb1.p[0] = *(const h8*)(vr + sw1 * 16);
      vb1.p[1] = *(const h8*)(vr + sw1 * 16 + 8);
      oacc[j] = __builtin_amdgcn_wmma_f32_16x16x32_f16(false, pa0.v, false, vb0.v, (short)0, oacc[j], false, false);
      oacc[j] = __builtin_amdgcn_wmma_f32_16x16x32_f16(false, pa1.v, false, vb1.v, (short)0, oacc[j], false, false);
    }
  }

  // ---- write O (C layout: lane holds (M = r + hi*8, N = lh)) ----
  {
    float* ob = Out + (rowbase + q0 + wave * 16 + hi * 8) * (size_t)D_ + lh;
#pragma unroll
    for (int j = 0; j < 4; ++j)
#pragma unroll
      for (int r = 0; r < 8; ++r)
        __builtin_nontemporal_store(oacc[j][r], ob + (size_t)r * D_ + j * 16);
  }
}

extern "C" void kernel_launch(void* const* d_in, const int* in_sizes, int n_in,
                              void* d_out, int out_size, void* d_ws, size_t ws_size,
                              hipStream_t stream) {
  (void)in_sizes; (void)n_in; (void)out_size; (void)d_ws; (void)ws_size;
  const float* Q   = (const float*)d_in[0];
  const float* K   = (const float*)d_in[1];
  const float* V   = (const float*)d_in[2];
  // d_in[3] (p_q) and d_in[4] (p_k) are unused by the reference
  const int*   Msk = (const int*)d_in[5];
  float* Out    = (float*)d_out;
  float* Scores = Out + (size_t)B_ * H_ * L_ * D_;   // tuple: (out, scores) concatenated flat

  dim3 grid(L_ / QT, B_ * H_);
  dim3 block(256);
  hipLaunchKernelGGL(PointSAMAttention_40973988004763_kernel, grid, block, 0, stream,
                     Q, K, V, Msk, Out, Scores);
}
